// ExtraMSABlock_16054587752606
// MI455X (gfx1250) — matrix operand order repaired and checked by reading, not compiled, not run on hardware
//
#include <hip/hip_runtime.h>
#include <math.h>

typedef _Float16 v16h __attribute__((ext_vector_type(16)));
typedef _Float16 v8h  __attribute__((ext_vector_type(8)));
typedef float    v8f  __attribute__((ext_vector_type(8)));

#define S_DIM 128
#define L_DIM 256
#define CMD   64
#define CZD   128
#define SLTOT (S_DIM * L_DIM)   /* 32768 */
#define LLTOT (L_DIM * L_DIM)   /* 65536 */

static __device__ __forceinline__ v16h cat8(v8h a, v8h b) {
  return __builtin_shufflevector(a, b, 0,1,2,3,4,5,6,7,8,9,10,11,12,13,14,15);
}
static __device__ __forceinline__ v8f wmma16(v16h a, v16h b, v8f c) {
  return __builtin_amdgcn_wmma_f32_16x16x32_f16(false, a, false, b, (short)0, c, false, false);
}
static __device__ __forceinline__ float sigm(float x) { return 1.f / (1.f + __expf(-x)); }

// ---------------------------------------------------------------- layernorm
__global__ void ln_kernel(const float* __restrict__ x, const float* __restrict__ gw,
                          const float* __restrict__ bw, float* __restrict__ y,
                          int rows, int C) {
  int wid = threadIdx.x >> 5, lane = threadIdx.x & 31;
  int row = blockIdx.x * 8 + wid;
  if (row >= rows) return;
  const float* xr = x + (long)row * C;
  float s1 = 0.f, s2 = 0.f;
  for (int c = lane; c < C; c += 32) { float v = xr[c]; s1 += v; s2 += v * v; }
  for (int o = 16; o > 0; o >>= 1) { s1 += __shfl_xor(s1, o, 32); s2 += __shfl_xor(s2, o, 32); }
  float mu = s1 / C;
  float var = s2 / C - mu * mu;
  float rstd = rsqrtf(var + 1e-5f);
  float* yr = y + (long)row * C;
  for (int c = lane; c < C; c += 32) yr[c] = (xr[c] - mu) * rstd * gw[c] + bw[c];
}

// ------------------------------------------------- generic WMMA GEMM + epilogues
// C[M,N] = epi(A[M,K] @ B[K,N]); fp32->f16 staging; double-buffered LDS.
// epi: 0 acc*scale+bias | 1 relu | 2 sigmoid | 3 resid+gate*(acc+b) | 4 resid+acc+b
//      5 OPM scatter    | 6 resid+(acc+b)/(rowv[m]+1e-3) | 7 (acc+b)*rowv[m]
struct GemmArgs {
  const float* A; long lda;
  const float* B; long ldb;
  float* C; long ldc;
  const float* bias; const float* gate; const float* resid; const float* rowv;
  float scale; int M, N, K, epi;
};

template <int TA>
__global__ __launch_bounds__(256) void gemm_kernel(GemmArgs g) {
  __shared__ _Float16 As[2][64][40];
  __shared__ _Float16 Bs[2][64][40];
  int tid = threadIdx.x, lane = tid & 31, wid = tid >> 5;
  int ln15 = lane & 15, lhalf = lane >> 4;
  int bm = blockIdx.y * 64, bn = blockIdx.x * 64;
  int wm = wid & 3, wn = wid >> 2;
  v8f acc[2] = {};
  const int rA = tid >> 2, c8 = (tid & 3) * 8;  // row-major staging mapping
  const int kB = tid >> 3, n8 = (tid & 7) * 8;  // transpose staging mapping
  const bool fullM = (bm + 64 <= g.M);
  const bool fullN = (bn + 64 <= g.N);
  float ra[8], rb[8];

  auto fetch = [&](int k0) {
    const bool fullK = (k0 + 32 <= g.K);
    if (TA == 0) {  // A row-major: contiguous 8 floats per thread
      if (fullK && fullM) {
        const float* src = g.A + (long)(bm + rA) * g.lda + k0 + c8;
        float4 f0 = *(const float4*)src;
        float4 f1 = *(const float4*)(src + 4);
        ra[0] = f0.x; ra[1] = f0.y; ra[2] = f0.z; ra[3] = f0.w;
        ra[4] = f1.x; ra[5] = f1.y; ra[6] = f1.z; ra[7] = f1.w;
        if (k0 + 64 < g.K) __builtin_prefetch(src + 64, 0, 1);
      } else {
        int row = bm + rA;
#pragma unroll
        for (int j = 0; j < 8; ++j) {
          int kk = k0 + c8 + j;
          ra[j] = (row < g.M && kk < g.K) ? g.A[(long)row * g.lda + kk] : 0.f;
        }
      }
    } else {  // A^T: contiguous along M, scatter into LDS
      int kk = k0 + kB;
      if (fullK && fullM) {
        const float* src = g.A + (long)kk * g.lda + bm + n8;
        float4 f0 = *(const float4*)src;
        float4 f1 = *(const float4*)(src + 4);
        ra[0] = f0.x; ra[1] = f0.y; ra[2] = f0.z; ra[3] = f0.w;
        ra[4] = f1.x; ra[5] = f1.y; ra[6] = f1.z; ra[7] = f1.w;
      } else {
#pragma unroll
        for (int j = 0; j < 8; ++j) {
          int col = bm + n8 + j;
          ra[j] = (kk < g.K && col < g.M) ? g.A[(long)kk * g.lda + col] : 0.f;
        }
      }
    }
    {  // B [K,N]: contiguous along N, scatter into LDS [n][k]
      int kk = k0 + kB;
      if (fullK && fullN) {
        const float* src = g.B + (long)kk * g.ldb + bn + n8;
        float4 f0 = *(const float4*)src;
        float4 f1 = *(const float4*)(src + 4);
        rb[0] = f0.x; rb[1] = f0.y; rb[2] = f0.z; rb[3] = f0.w;
        rb[4] = f1.x; rb[5] = f1.y; rb[6] = f1.z; rb[7] = f1.w;
      } else {
#pragma unroll
        for (int j = 0; j < 8; ++j) {
          int col = bn + n8 + j;
          rb[j] = (kk < g.K && col < g.N) ? g.B[(long)kk * g.ldb + col] : 0.f;
        }
      }
    }
  };
  auto commit = [&](int buf) {
    if (TA == 0) {
      v8h hv;
#pragma unroll
      for (int j = 0; j < 8; ++j) hv[j] = (_Float16)ra[j];
      *(v8h*)&As[buf][rA][c8] = hv;
    } else {
#pragma unroll
      for (int j = 0; j < 8; ++j) As[buf][n8 + j][kB] = (_Float16)ra[j];
    }
#pragma unroll
    for (int j = 0; j < 8; ++j) Bs[buf][n8 + j][kB] = (_Float16)rb[j];
  };

  fetch(0);
  commit(0);
  __syncthreads();
  int cur = 0;
  for (int k0 = 0; k0 < g.K; k0 += 32) {
    const bool hasNext = (k0 + 32) < g.K;
    if (hasNext) fetch(k0 + 32);  // global loads overlap with WMMA below
    v16h af;
    {
      int row = wm * 16 + ln15;
      af = cat8(*(const v8h*)&As[cur][row][lhalf * 8], *(const v8h*)&As[cur][row][16 + lhalf * 8]);
    }
#pragma unroll
    for (int nt = 0; nt < 2; ++nt) {
      int nrow = wn * 32 + nt * 16 + ln15;
      v16h bf = cat8(*(const v8h*)&Bs[cur][nrow][lhalf * 8],
                     *(const v8h*)&Bs[cur][nrow][16 + lhalf * 8]);
      acc[nt] = wmma16(af, bf, acc[nt]);
    }
    if (hasNext) {
      commit(cur ^ 1);
      __syncthreads();
      cur ^= 1;
    }
  }
  const bool guard = !(fullM && fullN);
#pragma unroll
  for (int nt = 0; nt < 2; ++nt)
#pragma unroll
    for (int rr = 0; rr < 8; ++rr) {
      int mg = bm + wm * 16 + 8 * lhalf + rr;
      int ng = bn + wn * 32 + nt * 16 + ln15;
      if (guard && (mg >= g.M || ng >= g.N)) continue;
      float acv = acc[nt][rr];
      float bsv = g.bias ? g.bias[ng] : 0.f;
      long cidx = (long)mg * g.ldc + ng;
      float outv;
      switch (g.epi) {
        case 0: outv = acv * g.scale + bsv; break;
        case 1: { float t = acv + bsv; outv = t > 0.f ? t : 0.f; } break;
        case 2: outv = sigm(acv + bsv); break;
        case 3: outv = g.resid[cidx] + g.gate[cidx] * (acv + bsv); break;
        case 4: outv = g.resid[cidx] + acv + bsv; break;
        case 5: cidx = ((long)(mg >> 5) * L_DIM + (ng >> 5)) * 1024 + (mg & 31) * 32 + (ng & 31);
                outv = acv; break;
        case 6: outv = g.resid[cidx] + (acv + bsv) / (g.rowv[mg] + 1e-3f); break;
        default: outv = (acv + bsv) * g.rowv[mg]; break;  // epi 7
      }
      g.C[cidx] = outv;
    }
}

// --------------------------------- triangle einsum: x[i,j,c] = sum_k a(i,k,c)*b(j,k,c)
// incoming==0: a[i,k,c], b[j,k,c]   (ikc,jkc->ijc)
// incoming==1: a[k,i,c], b[k,j,c]   (kic,kjc->ijc)
template <int INC>
__global__ __launch_bounds__(256) void tri_einsum_kernel(const float* __restrict__ a,
                                                         const float* __restrict__ b,
                                                         float* __restrict__ x) {
  __shared__ _Float16 As[2][64][40];
  __shared__ _Float16 Bs[2][64][40];
  int tid = threadIdx.x, lane = tid & 31, wid = tid >> 5;
  int ln15 = lane & 15, lhalf = lane >> 4;
  int c = blockIdx.z;
  int i0 = blockIdx.y * 64, j0 = blockIdx.x * 64;
  int wm = wid & 3, wn = wid >> 2;
  v8f acc[2] = {};
  const int rA = tid >> 2, c8 = (tid & 3) * 8;
  const long st = INC ? (long)L_DIM * CZD : (long)CZD;
  float ta[8], tb[8];

  auto fetch = [&](int k0) {
    const float *pa, *pb;
    if (INC) {
      pa = a + ((long)(k0 + c8) * L_DIM + (i0 + rA)) * CZD + c;
      pb = b + ((long)(k0 + c8) * L_DIM + (j0 + rA)) * CZD + c;
    } else {
      pa = a + ((long)(i0 + rA) * L_DIM + k0 + c8) * CZD + c;
      pb = b + ((long)(j0 + rA) * L_DIM + k0 + c8) * CZD + c;
    }
#pragma unroll
    for (int j = 0; j < 8; ++j) { ta[j] = pa[(long)j * st]; tb[j] = pb[(long)j * st]; }
  };
  auto commit = [&](int buf) {
#pragma unroll
    for (int j = 0; j < 8; ++j) {
      As[buf][rA][c8 + j] = (_Float16)ta[j];
      Bs[buf][rA][c8 + j] = (_Float16)tb[j];
    }
  };

  fetch(0);
  commit(0);
  __syncthreads();
  int cur = 0;
  for (int k0 = 0; k0 < L_DIM; k0 += 32) {
    const bool hasNext = (k0 + 32) < L_DIM;
    if (hasNext) fetch(k0 + 32);
    v16h af;
    {
      int row = wm * 16 + ln15;
      af = cat8(*(const v8h*)&As[cur][row][lhalf * 8], *(const v8h*)&As[cur][row][16 + lhalf * 8]);
    }
#pragma unroll
    for (int nt = 0; nt < 2; ++nt) {
      int nrow = wn * 32 + nt * 16 + ln15;
      v16h bf = cat8(*(const v8h*)&Bs[cur][nrow][lhalf * 8],
                     *(const v8h*)&Bs[cur][nrow][16 + lhalf * 8]);
      acc[nt] = wmma16(af, bf, acc[nt]);
    }
    if (hasNext) {
      commit(cur ^ 1);
      __syncthreads();
      cur ^= 1;
    }
  }
#pragma unroll
  for (int nt = 0; nt < 2; ++nt)
#pragma unroll
    for (int rr = 0; rr < 8; ++rr) {
      int ig = i0 + wm * 16 + 8 * lhalf + rr;
      int jg = j0 + wn * 32 + nt * 16 + ln15;
      x[((long)ig * L_DIM + jg) * CZD + c] = acc[nt][rr];
    }
}

// --------------------------- fused attention (flash-style, 32 queries per block)
// scores = q.k^T + pb[q,k,h] + (mask[b,k]-1)*1e9 ; softmax over k ; out = gate * (P @ V)
// D = head dim (8 for MSA row-attn, 32 for triangle-attn). Lq = Lk = 256.
template <int D>
__global__ __launch_bounds__(256) void attn_kernel(const float* __restrict__ q,
                                                   const float* __restrict__ k,
                                                   const float* __restrict__ v,
                                                   const float* __restrict__ gate,
                                                   const float* __restrict__ pb,
                                                   const float* __restrict__ mask,
                                                   float* __restrict__ out, int H) {
  const int Lk = 256;
  const int HD = H * D;
  int q0 = blockIdx.x * 32;
  int h = blockIdx.y;
  int bz = blockIdx.z;
  __shared__ _Float16 Kt[256][40];   // [key][d]   (d padded to 32)
  __shared__ _Float16 Vt[32][264];   // [d][key]
  __shared__ _Float16 Sp[32][264];   // scores -> probabilities
  int tid = threadIdx.x, lane = tid & 31, wid = tid >> 5;
  int ln15 = lane & 15, lhalf = lane >> 4;
  {  // stage K (one key row per thread), vectorized
    const float* kr = k + ((long)(bz * Lk + tid)) * HD + h * D;
    float t[32];
#pragma unroll
    for (int i = 0; i < D / 4; ++i) {
      float4 f = *(const float4*)(kr + 4 * i);
      t[4 * i + 0] = f.x; t[4 * i + 1] = f.y; t[4 * i + 2] = f.z; t[4 * i + 3] = f.w;
    }
#pragma unroll
    for (int d = D; d < 32; ++d) t[d] = 0.f;
#pragma unroll
    for (int i = 0; i < 4; ++i) {
      v8h hv;
#pragma unroll
      for (int j = 0; j < 8; ++j) hv[j] = (_Float16)t[8 * i + j];
      *(v8h*)&Kt[tid][8 * i] = hv;
    }
  }
  {  // stage V transposed: Vt[d][kk], batched strided gather
    int d = tid & 31;
    int kk0 = (tid >> 5) * 32;
    float t[32];
    const float* vbase = v + ((long)(bz * Lk + kk0)) * HD + h * D + d;
#pragma unroll
    for (int j = 0; j < 32; ++j) t[j] = (d < D) ? vbase[(long)j * HD] : 0.f;
#pragma unroll
    for (int i = 0; i < 4; ++i) {
      v8h hv;
#pragma unroll
      for (int j = 0; j < 8; ++j) hv[j] = (_Float16)t[8 * i + j];
      *(v8h*)&Vt[d][kk0 + 8 * i] = hv;
    }
  }
  // Q fragments straight from global (2 m-tiles of 16 rows)
  v16h qa[2];
#pragma unroll
  for (int mt = 0; mt < 2; ++mt) {
    v16h af = {};
    const float* qr = q + ((long)(bz * 256 + q0 + mt * 16 + ln15)) * HD + h * D;
    if (D == 32) {
      float4 a0 = *(const float4*)(qr + lhalf * 8);
      float4 a1 = *(const float4*)(qr + lhalf * 8 + 4);
      float4 b0 = *(const float4*)(qr + 16 + lhalf * 8);
      float4 b1 = *(const float4*)(qr + 16 + lhalf * 8 + 4);
      af[0] = (_Float16)a0.x; af[1] = (_Float16)a0.y; af[2] = (_Float16)a0.z; af[3] = (_Float16)a0.w;
      af[4] = (_Float16)a1.x; af[5] = (_Float16)a1.y; af[6] = (_Float16)a1.z; af[7] = (_Float16)a1.w;
      af[8] = (_Float16)b0.x; af[9] = (_Float16)b0.y; af[10] = (_Float16)b0.z; af[11] = (_Float16)b0.w;
      af[12] = (_Float16)b1.x; af[13] = (_Float16)b1.y; af[14] = (_Float16)b1.z; af[15] = (_Float16)b1.w;
    } else {  // D == 8: only kh==0 lanes carry data (d = 0..7)
      if (lhalf == 0) {
        float4 a0 = *(const float4*)qr;
        float4 a1 = *(const float4*)(qr + 4);
        af[0] = (_Float16)a0.x; af[1] = (_Float16)a0.y; af[2] = (_Float16)a0.z; af[3] = (_Float16)a0.w;
        af[4] = (_Float16)a1.x; af[5] = (_Float16)a1.y; af[6] = (_Float16)a1.z; af[7] = (_Float16)a1.w;
      }
    }
    qa[mt] = af;
  }
  __syncthreads();
  // ---- scores: each wave owns 2 key-tiles x both query-tiles
  for (int nt = wid * 2; nt < wid * 2 + 2; ++nt) {
    int nr = nt * 16 + ln15;
    v16h bf = cat8(*(const v8h*)&Kt[nr][lhalf * 8], *(const v8h*)&Kt[nr][16 + lhalf * 8]);
#pragma unroll
    for (int mt = 0; mt < 2; ++mt) {
      v8f acc = {};
      acc = wmma16(qa[mt], bf, acc);
#pragma unroll
      for (int rr = 0; rr < 8; ++rr) {
        int mm = mt * 16 + 8 * lhalf + rr;
        int nn = nt * 16 + ln15;
        float sv = acc[rr] + pb[((long)(q0 + mm) * Lk + nn) * H + h]
                 + (mask[(long)bz * Lk + nn] - 1.f) * 1e9f;
        Sp[mm][nn] = (_Float16)sv;
      }
    }
  }
  __syncthreads();
  // ---- softmax per row (wave per row group; lane covers 8 contiguous cols)
  for (int rr = wid; rr < 32; rr += 8) {
    float e[8];
    float mx = -1e30f;
#pragma unroll
    for (int j = 0; j < 8; ++j) { e[j] = (float)Sp[rr][lane * 8 + j]; mx = fmaxf(mx, e[j]); }
    for (int o = 16; o > 0; o >>= 1) mx = fmaxf(mx, __shfl_xor(mx, o, 32));
    float sm = 0.f;
#pragma unroll
    for (int j = 0; j < 8; ++j) { e[j] = __expf(e[j] - mx); sm += e[j]; }
    for (int o = 16; o > 0; o >>= 1) sm += __shfl_xor(sm, o, 32);
    float inv = 1.f / sm;
#pragma unroll
    for (int j = 0; j < 8; ++j) Sp[rr][lane * 8 + j] = (_Float16)(e[j] * inv);
  }
  __syncthreads();
  // ---- O = P @ V
  const int NT = (D + 15) / 16;
  int combos = 2 * NT;
  if (wid < combos) {
    int mt = wid & 1, nt = wid >> 1;
    v8f acc = {};
#pragma unroll
    for (int ks = 0; ks < 8; ++ks) {
      int row = mt * 16 + ln15;
      v16h af = cat8(*(const v8h*)&Sp[row][ks * 32 + lhalf * 8],
                     *(const v8h*)&Sp[row][ks * 32 + 16 + lhalf * 8]);
      int drow = nt * 16 + ln15;
      v16h bf = cat8(*(const v8h*)&Vt[drow][ks * 32 + lhalf * 8],
                     *(const v8h*)&Vt[drow][ks * 32 + 16 + lhalf * 8]);
      acc = wmma16(af, bf, acc);
    }
#pragma unroll
    for (int rr = 0; rr < 8; ++rr) {
      int mm = mt * 16 + 8 * lhalf + rr;
      int dd = nt * 16 + ln15;
      if (dd < D) {
        long oi = ((long)(bz * 256 + q0 + mm)) * HD + h * D + dd;
        out[oi] = gate[oi] * acc[rr];
      }
    }
  }
}

// ----------------------------------------------------------- small helpers (VALU)
__global__ void proj_small_kernel(const float* __restrict__ x, const float* __restrict__ w,
                                  const float* __restrict__ bias, float* __restrict__ out,
                                  int M, int N, int K) {
  long idx = (long)blockIdx.x * blockDim.x + threadIdx.x;
  if (idx >= (long)M * N) return;
  int m = (int)(idx / N), n = (int)(idx % N);
  float acc = bias ? bias[n] : 0.f;
  const float* xr = x + (long)m * K;
#pragma unroll 8
  for (int kk = 0; kk < K; ++kk) acc += xr[kk] * w[(long)kk * N + n];
  out[idx] = acc;
}

__global__ void mul3_kernel(const float4* __restrict__ a, const float4* __restrict__ b,
                            const float* __restrict__ mask, float4* __restrict__ out,
                            long n4, int C4) {
  long idx = (long)blockIdx.x * blockDim.x + threadIdx.x;
  if (idx >= n4) return;
  float mk = mask[idx / C4];
  float4 av = a[idx], bv = b[idx];
  float4 ov;
  ov.x = av.x * bv.x * mk; ov.y = av.y * bv.y * mk;
  ov.z = av.z * bv.z * mk; ov.w = av.w * bv.w * mk;
  out[idx] = ov;
}

__global__ void transposeLLC_kernel(const float* __restrict__ in, float* __restrict__ out, int C) {
  long idx = (long)blockIdx.x * blockDim.x + threadIdx.x;
  long total = (long)LLTOT * C;
  if (idx >= total) return;
  int c = (int)(idx % C);
  long ij = idx / C;
  int j = (int)(ij & 255), i = (int)(ij >> 8);
  out[(((long)j << 8) + i) * C + c] = in[idx];
}

__global__ void transposeLLC4_kernel(const float4* __restrict__ in, float4* __restrict__ out,
                                     int C4) {
  long idx = (long)blockIdx.x * blockDim.x + threadIdx.x;
  long total = (long)LLTOT * C4;
  if (idx >= total) return;
  int c = (int)(idx % C4);
  long ij = idx / C4;
  int j = (int)(ij & 255), i = (int)(ij >> 8);
  out[(((long)j << 8) + i) * C4 + c] = in[idx];
}

__global__ void qavg_kernel(const float* __restrict__ mln, const float* __restrict__ mask,
                            float* __restrict__ qavg) {
  int idx = blockIdx.x * blockDim.x + threadIdx.x;
  if (idx >= L_DIM * CMD) return;
  int l = idx >> 6, c = idx & 63;
  float num = 0.f, den = 1e-10f;
#pragma unroll 8
  for (int s = 0; s < S_DIM; ++s) {
    float mk = mask[(long)s * L_DIM + l];
    num += mk * mln[((long)s * L_DIM + l) * CMD + c];
    den += mk;
  }
  qavg[idx] = num / den;
}

__global__ void colattn_kernel(const float* __restrict__ qb, const float* __restrict__ kb,
                               const float* __restrict__ vb, const float* __restrict__ mask,
                               float* __restrict__ o2) {
  int l = blockIdx.x, h = blockIdx.y;
  int s = threadIdx.x;  // 128 threads
  __shared__ float sc[S_DIM];
  float dot = 0.f;
#pragma unroll
  for (int d = 0; d < 8; ++d) dot += qb[l * CMD + h * 8 + d] * kb[((long)s * L_DIM + l) * 8 + d];
  dot += (mask[(long)s * L_DIM + l] - 1.f) * 1e9f;
  sc[s] = dot;
  __syncthreads();
  float mx = -1e30f;
  for (int t = 0; t < S_DIM; ++t) mx = fmaxf(mx, sc[t]);
  float e = __expf(dot - mx);
  __syncthreads();
  sc[s] = e;
  __syncthreads();
  float sm = 0.f;
  for (int t = 0; t < S_DIM; ++t) sm += sc[t];
  if (s < 8) {
    float acc = 0.f;
    for (int t = 0; t < S_DIM; ++t) acc += sc[t] * vb[((long)t * L_DIM + l) * 8 + s];
    o2[l * CMD + h * 8 + s] = acc / sm;
  }
}

__global__ void gatebcast_kernel(const float* __restrict__ g, const float* __restrict__ o2,
                                 float* __restrict__ out) {
  long idx = (long)blockIdx.x * blockDim.x + threadIdx.x;
  if (idx >= (long)SLTOT * CMD) return;
  long sl = idx >> 6;
  int l = (int)(sl & 255);
  int c = (int)(idx & 63);
  out[idx] = g[idx] * o2[l * CMD + c];
}

// -------------------------------------------------------------------- host side
static void gemm(hipStream_t st, const float* A, long lda, int tA, const float* B, long ldb,
                 int tB, float* C, long ldc, const float* bias, const float* gate,
                 const float* resid, const float* rowv, float scale, int M, int N, int K,
                 int epi) {
  (void)tB;
  GemmArgs ga{A, lda, B, ldb, C, ldc, bias, gate, resid, rowv, scale, M, N, K, epi};
  dim3 grid((N + 63) / 64, (M + 63) / 64);
  if (tA)
    gemm_kernel<1><<<grid, 256, 0, st>>>(ga);
  else
    gemm_kernel<0><<<grid, 256, 0, st>>>(ga);
}
static void ln(hipStream_t st, const float* x, const float* g, const float* b, float* y,
               int rows, int C) {
  ln_kernel<<<(rows + 7) / 8, 256, 0, st>>>(x, g, b, y, rows, C);
}
static void proj_small(hipStream_t st, const float* x, const float* w, const float* bias,
                       float* out, int M, int N, int K) {
  long n = (long)M * N;
  proj_small_kernel<<<(int)((n + 255) / 256), 256, 0, st>>>(x, w, bias, out, M, N, K);
}

struct TriMultP { const float *ln_g,*ln_b,*ag_w,*ag_b,*ap_w,*ap_b,*bg_w,*bg_b,*bp_w,*bp_b,*ln2_g,*ln2_b,*g_w,*g_b,*o_w,*o_b; };
struct TriAttP  { const float *ln_g,*ln_b,*b_w,*q_w,*k_w,*v_w,*g_w,*g_b,*o_w,*o_b; };

extern "C" void kernel_launch(void* const* d_in, const int* in_sizes, int n_in, void* d_out,
                              int out_size, void* d_ws, size_t ws_size, hipStream_t stream) {
  (void)in_sizes; (void)n_in; (void)out_size; (void)ws_size;
  const float* m_in = (const float*)d_in[0];
  const float* z_in = (const float*)d_in[1];
  const float* msa_mask = (const float*)d_in[2];
  const float* pair_mask = (const float*)d_in[3];

  int pi = 4;
  auto NX = [&]() { return (const float*)d_in[pi++]; };
  // row
  const float *r_ln_m_g = NX(), *r_ln_m_b = NX(), *r_ln_z_g = NX(), *r_ln_z_b = NX();
  const float *r_b_w = NX(), *r_q_w = NX(), *r_k_w = NX(), *r_v_w = NX();
  const float *r_g_w = NX(), *r_g_b = NX(), *r_o_w = NX(), *r_o_b = NX();
  // col
  const float *c_ln_g = NX(), *c_ln_b = NX(), *c_q_w = NX(), *c_k_w = NX(), *c_v_w = NX();
  const float *c_g_w = NX(), *c_g_b = NX(), *c_o_w = NX(), *c_o_b = NX();
  // mt
  const float *mt_ln_g = NX(), *mt_ln_b = NX(), *mt_w1 = NX(), *mt_b1 = NX(), *mt_w2 = NX(),
              *mt_b2 = NX();
  // opm
  const float *op_ln_g = NX(), *op_ln_b = NX(), *op_a_w = NX(), *op_a_b = NX(), *op_b_w = NX(),
              *op_b_b = NX(), *op_o_w = NX(), *op_o_b = NX();
  auto getTM = [&]() {
    TriMultP t;
    t.ln_g = NX(); t.ln_b = NX(); t.ag_w = NX(); t.ag_b = NX(); t.ap_w = NX(); t.ap_b = NX();
    t.bg_w = NX(); t.bg_b = NX(); t.bp_w = NX(); t.bp_b = NX(); t.ln2_g = NX(); t.ln2_b = NX();
    t.g_w = NX(); t.g_b = NX(); t.o_w = NX(); t.o_b = NX();
    return t;
  };
  TriMultP tmo = getTM(), tmi = getTM();
  auto getTA = [&]() {
    TriAttP t;
    t.ln_g = NX(); t.ln_b = NX(); t.b_w = NX(); t.q_w = NX(); t.k_w = NX(); t.v_w = NX();
    t.g_w = NX(); t.g_b = NX(); t.o_w = NX(); t.o_b = NX();
    return t;
  };
  TriAttP tas = getTA(), tae = getTA();
  const float *pt_ln_g = NX(), *pt_ln_b = NX(), *pt_w1 = NX(), *pt_b1 = NX(), *pt_w2 = NX(),
              *pt_b2 = NX();

  // ---------------- workspace layout
  float* w = (float*)d_ws;
  float* m_cur = w; w += (long)SLTOT * CMD;
  float* z_cur = w; w += (long)LLTOT * CZD;
  float* P[7];
  for (int i = 0; i < 7; ++i) { P[i] = w; w += (long)LLTOT * CZD; }
  float* pbrow = w; w += (long)LLTOT * 8;
  float* pbtri = w; w += (long)LLTOT * 4;
  float* normb = w; w += LLTOT;
  float* maskT = w; w += LLTOT;
  float* qavg = w; w += L_DIM * CMD;
  float* qcol = w; w += L_DIM * CMD;
  float* kcol = w; w += (long)SLTOT * 8;
  float* vcol = w; w += (long)SLTOT * 8;
  float* obuf2 = w; w += L_DIM * CMD;

  const float rsq8 = 0.35355339059327373f;   // 1/sqrt(8)
  const float rsq32 = 0.17677669529663687f;  // 1/sqrt(32)

  // ================= MSA row attention (pair-biased, gated) =================
  ln(stream, m_in, r_ln_m_g, r_ln_m_b, P[0], SLTOT, CMD);
  ln(stream, z_in, r_ln_z_g, r_ln_z_b, P[1], LLTOT, CZD);
  proj_small(stream, P[1], r_b_w, nullptr, pbrow, LLTOT, 8, CZD);
  gemm(stream, P[0], CMD, 0, r_q_w, CMD, 0, P[2], CMD, nullptr, nullptr, nullptr, nullptr, rsq8, SLTOT, CMD, CMD, 0);
  gemm(stream, P[0], CMD, 0, r_k_w, CMD, 0, P[3], CMD, nullptr, nullptr, nullptr, nullptr, 1.f, SLTOT, CMD, CMD, 0);
  gemm(stream, P[0], CMD, 0, r_v_w, CMD, 0, P[4], CMD, nullptr, nullptr, nullptr, nullptr, 1.f, SLTOT, CMD, CMD, 0);
  gemm(stream, P[0], CMD, 0, r_g_w, CMD, 0, P[5], CMD, r_g_b, nullptr, nullptr, nullptr, 1.f, SLTOT, CMD, CMD, 2);
  attn_kernel<8><<<dim3(8, 8, S_DIM), 256, 0, stream>>>(P[2], P[3], P[4], P[5], pbrow, msa_mask, P[6], 8);
  gemm(stream, P[6], CMD, 0, r_o_w, CMD, 0, m_cur, CMD, r_o_b, nullptr, m_in, nullptr, 1.f, SLTOT, CMD, CMD, 4);

  // ================= MSA column global attention =================
  ln(stream, m_cur, c_ln_g, c_ln_b, P[0], SLTOT, CMD);
  qavg_kernel<<<(L_DIM * CMD + 255) / 256, 256, 0, stream>>>(P[0], msa_mask, qavg);
  gemm(stream, qavg, CMD, 0, c_q_w, CMD, 0, qcol, CMD, nullptr, nullptr, nullptr, nullptr, rsq8, L_DIM, CMD, CMD, 0);
  proj_small(stream, P[0], c_k_w, nullptr, kcol, SLTOT, 8, CMD);
  proj_small(stream, P[0], c_v_w, nullptr, vcol, SLTOT, 8, CMD);
  colattn_kernel<<<dim3(L_DIM, 8), S_DIM, 0, stream>>>(qcol, kcol, vcol, msa_mask, obuf2);
  gemm(stream, P[0], CMD, 0, c_g_w, CMD, 0, P[1], CMD, c_g_b, nullptr, nullptr, nullptr, 1.f, SLTOT, CMD, CMD, 2);
  gatebcast_kernel<<<(int)(((long)SLTOT * CMD + 255) / 256), 256, 0, stream>>>(P[1], obuf2, P[2]);
  gemm(stream, P[2], CMD, 0, c_o_w, CMD, 0, m_cur, CMD, c_o_b, nullptr, m_cur, nullptr, 1.f, SLTOT, CMD, CMD, 4);

  // ================= MSA transition =================
  ln(stream, m_cur, mt_ln_g, mt_ln_b, P[0], SLTOT, CMD);
  gemm(stream, P[0], CMD, 0, mt_w1, 4 * CMD, 0, P[1], 4 * CMD, mt_b1, nullptr, nullptr, nullptr, 1.f, SLTOT, 4 * CMD, CMD, 1);
  gemm(stream, P[1], 4 * CMD, 0, mt_w2, CMD, 0, m_cur, CMD, mt_b2, nullptr, m_cur, nullptr, 1.f, SLTOT, CMD, 4 * CMD, 4);

  // ================= outer product mean -> z =================
  ln(stream, m_cur, op_ln_g, op_ln_b, P[0], SLTOT, CMD);
  gemm(stream, P[0], CMD, 0, op_a_w, 32, 0, P[1], 32, op_a_b, nullptr, nullptr, msa_mask, 1.f, SLTOT, 32, CMD, 7);
  gemm(stream, P[0], CMD, 0, op_b_w, 32, 0, P[2], 32, op_b_b, nullptr, nullptr, msa_mask, 1.f, SLTOT, 32, CMD, 7);
  gemm(stream, msa_mask, L_DIM, 1, msa_mask, L_DIM, 0, normb, L_DIM, nullptr, nullptr, nullptr, nullptr, 1.f, L_DIM, L_DIM, S_DIM, 0);
  for (int ci = 0; ci < 8; ++ci) {  // 32 i-rows per chunk
    gemm(stream, P[1] + (long)ci * 1024, 8192, 1 /*transA*/, P[2], 8192, 0, P[3], 8192,
         nullptr, nullptr, nullptr, nullptr, 1.f, 1024, 8192, S_DIM, 5);
    long zoff = (long)ci * 32 * L_DIM * CZD;
    gemm(stream, P[3], 1024, 0, op_o_w, CZD, 0, z_cur + zoff, CZD, op_o_b, nullptr,
         z_in + zoff, normb + (long)ci * 32 * L_DIM, 1.f, 32 * L_DIM, CZD, 1024, 6);
  }

  // ================= triangle multiplicative updates =================
  auto runTriMult = [&](const TriMultP& t, int incoming) {
    ln(stream, z_cur, t.ln_g, t.ln_b, P[0], LLTOT, CZD);
    gemm(stream, P[0], CZD, 0, t.ag_w, CZD, 0, P[1], CZD, t.ag_b, nullptr, nullptr, nullptr, 1.f, LLTOT, CZD, CZD, 2);
    gemm(stream, P[0], CZD, 0, t.ap_w, CZD, 0, P[2], CZD, t.ap_b, nullptr, nullptr, nullptr, 1.f, LLTOT, CZD, CZD, 0);
    mul3_kernel<<<(int)(((long)LLTOT * CZD / 4 + 255) / 256), 256, 0, stream>>>(
        (const float4*)P[1], (const float4*)P[2], pair_mask, (float4*)P[1],
        (long)LLTOT * CZD / 4, CZD / 4);
    gemm(stream, P[0], CZD, 0, t.bg_w, CZD, 0, P[2], CZD, t.bg_b, nullptr, nullptr, nullptr, 1.f, LLTOT, CZD, CZD, 2);
    gemm(stream, P[0], CZD, 0, t.bp_w, CZD, 0, P[4], CZD, t.bp_b, nullptr, nullptr, nullptr, 1.f, LLTOT, CZD, CZD, 0);
    mul3_kernel<<<(int)(((long)LLTOT * CZD / 4 + 255) / 256), 256, 0, stream>>>(
        (const float4*)P[2], (const float4*)P[4], pair_mask, (float4*)P[2],
        (long)LLTOT * CZD / 4, CZD / 4);
    if (incoming)
      tri_einsum_kernel<1><<<dim3(4, 4, CZD), 256, 0, stream>>>(P[1], P[2], P[3]);
    else
      tri_einsum_kernel<0><<<dim3(4, 4, CZD), 256, 0, stream>>>(P[1], P[2], P[3]);
    ln(stream, P[3], t.ln2_g, t.ln2_b, P[3], LLTOT, CZD);
    gemm(stream, P[0], CZD, 0, t.g_w, CZD, 0, P[4], CZD, t.g_b, nullptr, nullptr, nullptr, 1.f, LLTOT, CZD, CZD, 2);
    gemm(stream, P[3], CZD, 0, t.o_w, CZD, 0, z_cur, CZD, t.o_b, P[4], z_cur, nullptr, 1.f, LLTOT, CZD, CZD, 3);
  };
  runTriMult(tmo, 0);  // outgoing: ikc,jkc->ijc
  runTriMult(tmi, 1);  // incoming: kic,kjc->ijc

  // ================= triangle attention (start + end) =================
  auto runTriAtt = [&](const TriAttP& t, float* zbuf, const float* mk) {
    ln(stream, zbuf, t.ln_g, t.ln_b, P[0], LLTOT, CZD);
    proj_small(stream, P[0], t.b_w, nullptr, pbtri, LLTOT, 4, CZD);
    gemm(stream, P[0], CZD, 0, t.q_w, CZD, 0, P[1], CZD, nullptr, nullptr, nullptr, nullptr, rsq32, LLTOT, CZD, CZD, 0);
    gemm(stream, P[0], CZD, 0, t.k_w, CZD, 0, P[2], CZD, nullptr, nullptr, nullptr, nullptr, 1.f, LLTOT, CZD, CZD, 0);
    gemm(stream, P[0], CZD, 0, t.v_w, CZD, 0, P[3], CZD, nullptr, nullptr, nullptr, nullptr, 1.f, LLTOT, CZD, CZD, 0);
    gemm(stream, P[0], CZD, 0, t.g_w, CZD, 0, P[4], CZD, t.g_b, nullptr, nullptr, nullptr, 1.f, LLTOT, CZD, CZD, 2);
    attn_kernel<32><<<dim3(8, 4, L_DIM), 256, 0, stream>>>(P[1], P[2], P[3], P[4], pbtri, mk, P[5], 4);
    gemm(stream, P[5], CZD, 0, t.o_w, CZD, 0, zbuf, CZD, t.o_b, nullptr, zbuf, nullptr, 1.f, LLTOT, CZD, CZD, 4);
  };
  runTriAtt(tas, z_cur, pair_mask);
  // ending node: operate on transposed z
  transposeLLC_kernel<<<(LLTOT + 255) / 256, 256, 0, stream>>>(pair_mask, maskT, 1);
  transposeLLC4_kernel<<<(int)(((long)LLTOT * (CZD / 4) + 255) / 256), 256, 0, stream>>>(
      (const float4*)z_cur, (float4*)P[6], CZD / 4);
  runTriAtt(tae, P[6], maskT);
  transposeLLC4_kernel<<<(int)(((long)LLTOT * (CZD / 4) + 255) / 256), 256, 0, stream>>>(
      (const float4*)P[6], (float4*)z_cur, CZD / 4);

  // ================= pair transition (chunked over rows) =================
  ln(stream, z_cur, pt_ln_g, pt_ln_b, P[0], LLTOT, CZD);
  for (int ch = 0; ch < 4; ++ch) {
    long off = (long)ch * 16384;
    gemm(stream, P[0] + off * CZD, CZD, 0, pt_w1, 4 * CZD, 0, P[1], 4 * CZD, pt_b1, nullptr,
         nullptr, nullptr, 1.f, 16384, 4 * CZD, CZD, 1);
    gemm(stream, P[1], 4 * CZD, 0, pt_w2, CZD, 0, z_cur + off * CZD, CZD, pt_b2, nullptr,
         z_cur + off * CZD, nullptr, 1.f, 16384, CZD, 4 * CZD, 4);
  }

  // ================= outputs: (m, z) concatenated =================
  float* outp = (float*)d_out;
  hipMemcpyAsync(outp, m_cur, (size_t)SLTOT * CMD * sizeof(float), hipMemcpyDeviceToDevice, stream);
  hipMemcpyAsync(outp + (long)SLTOT * CMD, z_cur, (size_t)LLTOT * CZD * sizeof(float),
                 hipMemcpyDeviceToDevice, stream);
}
